// LCALayer_90494960926819
// MI455X (gfx1250) — matrix operand order repaired and checked by reading, not compiled, not run on hardware
//
#include <hip/hip_runtime.h>

// ---------------------------------------------------------------------------
// LCA layer on MI455X (gfx1250): bf16 WMMA (16x16x32, f32 accumulate)
// Wave tile 64x64 (16 WMMA tiles), block tile 128x256 (8 wave32s, 2x4).
// ---------------------------------------------------------------------------

typedef __attribute__((ext_vector_type(16))) __bf16 bf16x16;
typedef __attribute__((ext_vector_type(8)))  __bf16 bf16x8;
typedef __attribute__((ext_vector_type(8)))  float  floatx8;

#define LCA_DT_TAU 0.1f   // DT / TAU
#define LCA_LAM    0.1f

// ---- converters ------------------------------------------------------------

// W[1024][4096] f32 -> Wb bf16 (same layout) and Wtb bf16 [4096][1024]
__global__ void lca_cvt_w(const float* __restrict__ W,
                          __bf16* __restrict__ Wb,
                          __bf16* __restrict__ Wtb) {
  int i = blockIdx.x * blockDim.x + threadIdx.x;  // 0 .. 1024*4096
  int r = i >> 12;       // row  (d_model, 1024)
  int c = i & 4095;      // col  (d_lca, 4096)
  float v = W[i];
  Wb[i] = (__bf16)v;
  Wtb[(size_t)c * 1024 + r] = (__bf16)v;
}

__global__ void lca_cvt_x(const float* __restrict__ X, __bf16* __restrict__ Xb) {
  int i = blockIdx.x * blockDim.x + threadIdx.x;
  Xb[i] = (__bf16)X[i];
}

// ---- fused WMMA GEMM -------------------------------------------------------
// C[M x Nld] = A[M x K](bf16, row-major, lda) * B, where B is given by its
// transpose Bt (row-major, ldbt): B[k][n] == Bt[n][k].  So every B-fragment
// load is a contiguous row read (G is symmetric; Wt/W give the other cases).
//
// MODE 0: BO = bf16(C), zero diagonal            (G = Wt*W)
// MODE 1: F0 = C (b); F1 = 0.1*C (u1); BO = bf16(relu(u1-lam))  (a1)
// MODE 2: un = u + 0.1*(b - C - u); F1(u) in-place; BO = bf16(relu(un-lam))
// MODE 3: F0 = C (f32 output)                    (out = a*Wt)
template <int MODE>
__global__ void __launch_bounds__(256)
lca_gemm(const __bf16* __restrict__ A, const __bf16* __restrict__ Bt,
         int lda, int ldbt, int K, int Nld,
         float* __restrict__ F0, float* __restrict__ F1,
         __bf16* __restrict__ BO) {
  const int lane = threadIdx.x & 31;     // wave32
  const int wave = threadIdx.x >> 5;     // 8 waves / workgroup
  const int hl   = lane >> 4;            // half-wave select
  const int r    = lane & 15;
  const int wrow = wave & 1;             // 2 waves along M (2*64 = 128)
  const int wcol = wave >> 1;            // 4 waves along N (4*64 = 256)
  const int m0 = blockIdx.y * 128 + wrow * 64;
  const int n0 = blockIdx.x * 256 + wcol * 64;

  // A fragment base: this code's contiguous k order (matches B):
  //   lanes 0-15 : row M=r, elems = K 0..15
  //   lanes 16-31: row M=r, elems = K 16..31
  const __bf16* pA[4];
#pragma unroll
  for (int ms = 0; ms < 4; ++ms)
    pA[ms] = A + (size_t)(m0 + ms * 16 + r) * lda + hl * 16;

  // B fragment base: this code's contiguous k order (matches A):
  //   lanes 0-15 : col N=r, K 0..15 ; lanes 16-31: col N=r, K 16..31
  const __bf16* pB[4];
#pragma unroll
  for (int ns = 0; ns < 4; ++ns)
    pB[ns] = Bt + (size_t)(n0 + ns * 16 + r) * ldbt + hl * 16;

  floatx8 acc[4][4] = {};   // 128 f32 accumulators / lane

  for (int k0 = 0; k0 < K; k0 += 32) {
    union { bf16x16 v; bf16x8 h[2]; } aF[4];
    bf16x16 bF[4];
#pragma unroll
    for (int ms = 0; ms < 4; ++ms) {
      aF[ms].h[0] = *(const bf16x8*)(pA[ms] + k0);
      aF[ms].h[1] = *(const bf16x8*)(pA[ms] + k0 + 8);
    }
#pragma unroll
    for (int ns = 0; ns < 4; ++ns)
      bF[ns] = *(const bf16x16*)(pB[ns] + k0);

    if (k0 + 64 < K) {                   // one full cacheline ahead
      __builtin_prefetch(pA[0] + k0 + 64, 0, 1);
      __builtin_prefetch(pB[0] + k0 + 64, 0, 1);
    }

#pragma unroll
    for (int ms = 0; ms < 4; ++ms)
#pragma unroll
      for (int ns = 0; ns < 4; ++ns)
        acc[ms][ns] = __builtin_amdgcn_wmma_f32_16x16x32_bf16(
            false, aF[ms].v, false, bF[ns], (short)0, acc[ms][ns],
            false, false);
  }

  // Epilogue. C/D layout: element i -> M = 8*hl + i, N = r.
#pragma unroll
  for (int ms = 0; ms < 4; ++ms) {
#pragma unroll
    for (int ns = 0; ns < 4; ++ns) {
#pragma unroll
      for (int i = 0; i < 8; ++i) {
        int row = m0 + ms * 16 + hl * 8 + i;
        int col = n0 + ns * 16 + r;
        size_t idx = (size_t)row * Nld + col;
        float v = acc[ms][ns][i];
        if constexpr (MODE == 0) {
          BO[idx] = (row == col) ? (__bf16)0.0f : (__bf16)v;
        } else if constexpr (MODE == 1) {
          F0[idx] = v;                         // b
          float un = LCA_DT_TAU * v;           // u1 = 0.1*b
          F1[idx] = un;
          BO[idx] = (__bf16)fmaxf(un - LCA_LAM, 0.0f);
        } else if constexpr (MODE == 2) {
          float bv = F0[idx];
          float uv = F1[idx];
          float un = uv + LCA_DT_TAU * (bv - v - uv);
          F1[idx] = un;                        // u updated in place (tile-local)
          BO[idx] = (__bf16)fmaxf(un - LCA_LAM, 0.0f);
        } else {
          F0[idx] = v;                         // final f32 output
        }
      }
    }
  }
}

// ---------------------------------------------------------------------------

extern "C" void kernel_launch(void* const* d_in, const int* in_sizes, int n_in,
                              void* d_out, int out_size, void* d_ws, size_t ws_size,
                              hipStream_t stream) {
  (void)in_sizes; (void)n_in; (void)out_size; (void)ws_size;
  const float* x = (const float*)d_in[0];   // [4,2048,1024] -> [8192,1024]
  const float* W = (const float*)d_in[1];   // [1024,4096]
  float* out = (float*)d_out;               // [8192,1024]

  char* ws = (char*)d_ws;
  size_t off = 0;
  auto take = [&](size_t bytes) {
    char* p = ws + off;
    off += (bytes + 255) & ~(size_t)255;
    return p;
  };
  __bf16* Wb  = (__bf16*)take(1024ull * 4096 * 2);   // W bf16
  __bf16* Wtb = (__bf16*)take(4096ull * 1024 * 2);   // W^T bf16
  __bf16* Gb  = (__bf16*)take(4096ull * 4096 * 2);   // G bf16 (symmetric)
  __bf16* Xb  = (__bf16*)take(8192ull * 1024 * 2);   // x bf16
  float*  Bf  = (float*) take(8192ull * 4096 * 4);   // b  f32
  float*  Uf  = (float*) take(8192ull * 4096 * 4);   // u  f32 (in-place)
  __bf16* Aa0 = (__bf16*)take(8192ull * 4096 * 2);   // a ping
  __bf16* Aa1 = (__bf16*)take(8192ull * 4096 * 2);   // a pong
  __bf16* Aa[2] = {Aa0, Aa1};

  lca_cvt_w<<<(1024 * 4096) / 256, 256, 0, stream>>>(W, Wb, Wtb);
  lca_cvt_x<<<(8192 * 1024) / 256, 256, 0, stream>>>(x, Xb);

  // G = W^T W : M=N=4096, K=1024.  A = Wt rows; B-cols = Wt rows.
  lca_gemm<0><<<dim3(4096 / 256, 4096 / 128), 256, 0, stream>>>(
      Wtb, Wtb, 1024, 1024, 1024, 4096, nullptr, nullptr, Gb);

  // b = x @ W : M=8192, N=4096, K=1024.  B-cols of W = rows of Wt.
  // Fused: u1 = 0.1*b ; a1 = relu(u1 - lam).
  lca_gemm<1><<<dim3(4096 / 256, 8192 / 128), 256, 0, stream>>>(
      Xb, Wtb, 1024, 1024, 1024, 4096, Bf, Uf, Aa[0]);

  // 9 remaining LCA steps: C = a@G (G symmetric -> B-cols = G rows).
  int cur = 0;
  for (int s = 0; s < 9; ++s) {
    lca_gemm<2><<<dim3(4096 / 256, 8192 / 128), 256, 0, stream>>>(
        Aa[cur], Gb, 4096, 4096, 4096, 4096, Bf, Uf, Aa[1 - cur]);
    cur ^= 1;
  }

  // out = a @ W^T : M=8192, N=1024, K=4096.  B-cols of W^T = rows of W.
  lca_gemm<3><<<dim3(1024 / 256, 8192 / 128), 256, 0, stream>>>(
      Aa[cur], Wb, 4096, 4096, 4096, 1024, out, nullptr, nullptr);
}
